// GCCN_4492535791673
// MI455X (gfx1250) — matrix operand order repaired and checked by reading, not compiled, not run on hardware
//
#include <hip/hip_runtime.h>
#include <hip/hip_bf16.h>

#define H      128
#define LAYERS 3
#define COUT   16

typedef __bf16 bf16x16 __attribute__((ext_vector_type(16)));
typedef __bf16 bf16x8  __attribute__((ext_vector_type(8)));
typedef __bf16 bf16x4  __attribute__((ext_vector_type(4)));
typedef float  f32x8   __attribute__((ext_vector_type(8)));
typedef unsigned int u32x4 __attribute__((ext_vector_type(4)));
typedef int    i32x4   __attribute__((ext_vector_type(4)));
typedef int    i32x8   __attribute__((ext_vector_type(8)));

#define HAVE_TDM __has_builtin(__builtin_amdgcn_tensor_load_to_lds)

// WinT / WlT / WrT: [n=0..127][k stride 136] bf16 ; WoutT: [n=0..15][k stride 392]
#define BT_ELEMS (128 * 136)   // 17408 elems = 34816 bytes per 128x128 weight

#define CAT16(lo, hi)                                                                  \
    __builtin_shufflevector(lo, hi, 0, 1, 2, 3, 4, 5, 6, 7, 8, 9, 10, 11, 12, 13, 14, 15)

static __device__ __forceinline__ f32x8 wmma_bf16(bf16x16 a, bf16x16 b, f32x8 c) {
    return __builtin_amdgcn_wmma_f32_16x16x32_bf16(false, a, false, b, (short)0, c,
                                                   false, false);
}

static __device__ __forceinline__ bf16x4 cvt4(float4 v) {
    bf16x4 r;
    r[0] = (__bf16)v.x;
    r[1] = (__bf16)v.y;
    r[2] = (__bf16)v.z;
    r[3] = (__bf16)v.w;
    return r;
}

// A-frag: two contiguous 16B runs at k = g*8 and 16+g*8 (row stride 40 elems)
static __device__ __forceinline__ bf16x16 load_afrag(const __bf16* sA, int row, int g) {
    const __bf16* p = sA + row * 40 + g * 8;
    bf16x8 lo = *(const bf16x8*)p;
    bf16x8 hi = *(const bf16x8*)(p + 16);
    return CAT16(lo, hi);
}

// B-frag: two contiguous 16B runs at k = ko + g*8 and ko + 16 + g*8 in transposed staging
static __device__ __forceinline__ bf16x16 load_bfrag(const __bf16* sBt, int n, int ko,
                                                     int g, int kstride) {
    const __bf16* p = sBt + n * kstride + ko + g * 8;
    bf16x8 lo = *(const bf16x8*)p;
    bf16x8 hi = *(const bf16x8*)(p + 16);
    return CAT16(lo, hi);
}

#if HAVE_TDM
// One-shot 1-D TDM copy: global (flat, 8B elems) -> LDS.  ISA 08 D# layout.
static __device__ __forceinline__ void tdm_load(const __bf16* gsrc, __bf16* ldst,
                                                unsigned nbytes) {
    unsigned long long ga = (unsigned long long)(uintptr_t)gsrc;
    unsigned lds = (unsigned)(uintptr_t)ldst;   // LDS aperture: addr[31:0] = offset
    unsigned n8 = nbytes >> 3;                  // 8-byte elements (<= 65535)
    u32x4 g0;
    g0[0] = 1u;                                              // count=1, user mode
    g0[1] = lds;                                             // lds_addr
    g0[2] = (unsigned)ga;                                    // global_addr[31:0]
    g0[3] = (unsigned)((ga >> 32) & 0x1FFFFFFull) | (2u << 30);  // addr[56:32],type=2
    i32x8 g1;
    g1[0] = 0x00030000;                  // workgroup_mask=0, data_size=3 (8B)
    g1[1] = (int)(n8 << 16);             // tensor_dim0[15:0]
    g1[2] = (int)((n8 >> 16) & 0xFFFF) | (1 << 16);  // tensor_dim0[31:16], dim1=1
    g1[3] = (int)(n8 << 16);             // tile_dim0 = n8
    g1[4] = 1;                           // tile_dim1 = 1
    g1[5] = (int)n8;                     // tensor_dim0_stride lo
    g1[6] = 0;
    g1[7] = 0;
    i32x4 z = {0, 0, 0, 0};
#if defined(__clang_major__) && __clang_major__ >= 23
    i32x8 z8 = {0, 0, 0, 0, 0, 0, 0, 0};
    __builtin_amdgcn_tensor_load_to_lds(g0, g1, z, z, z8, 0);
#else
    __builtin_amdgcn_tensor_load_to_lds(g0, g1, z, z, 0);
#endif
}
#endif

// ---------------------------------------------------------------------------
// Weight prep: fp32 row-major [K][Ncols] -> bf16 transposed [n][kstride]
// ---------------------------------------------------------------------------
__global__ void k_transpose(const float* __restrict__ src, __bf16* __restrict__ dst,
                            int K, int Ncols, int kstride) {
    int i = blockIdx.x * blockDim.x + threadIdx.x;
    if (i >= K * Ncols) return;
    int k = i / Ncols, n = i % Ncols;
    dst[n * kstride + k] = (__bf16)src[(size_t)k * Ncols + n];
}

// ---------------------------------------------------------------------------
// Input projection: h0 = relu(x @ W_in + b_in)   [N,128]x[128,128]
// ---------------------------------------------------------------------------
__global__ __launch_bounds__(256) void k_in_proj(const float* __restrict__ x,
                                                 const __bf16* __restrict__ WinT,
                                                 const float* __restrict__ bin,
                                                 float* __restrict__ h0, int N) {
    __shared__ __bf16 sBt[BT_ELEMS];
    __shared__ __bf16 sA[2][32 * 40];
    const int t = threadIdx.x;
    const int w = t >> 5, lane = t & 31;
    const int row0 = blockIdx.x * 32;
    const int rm = (w >> 2) * 16, nb = (w & 3) * 32;
    const int g = lane >> 4, ln = lane & 15;
    const bool full = (row0 + 32 <= N);

    // stage B once: flat 16B-chunk copy of the pre-transposed bf16 image
    for (int i = t; i < BT_ELEMS / 8; i += 256)
        ((bf16x8*)sBt)[i] = ((const bf16x8*)WinT)[i];

    f32x8 acc0 = {}, acc1 = {};
    for (int kb = 0; kb < 4; ++kb) {
        {   // stage A (double buffered): one float4 + one 8B LDS store per thread
            int r = t >> 3, k4 = (t & 7) * 4;
            int gm = row0 + r;
            float4 v = make_float4(0.f, 0.f, 0.f, 0.f);
            if (gm < N) v = *(const float4*)(x + (size_t)gm * H + kb * 32 + k4);
            *(bf16x4*)&sA[kb & 1][r * 40 + k4] = cvt4(v);
        }
        __syncthreads();
        bf16x16 a  = load_afrag(sA[kb & 1], rm + ln, g);
        bf16x16 b0 = load_bfrag(sBt, nb + ln, kb * 32, g, 136);
        bf16x16 b1 = load_bfrag(sBt, nb + 16 + ln, kb * 32, g, 136);
        acc0 = wmma_bf16(a, b0, acc0);
        acc1 = wmma_bf16(a, b1, acc1);
    }
    float bv0 = bin[nb + ln], bv1 = bin[nb + 16 + ln];
    if (full) {
#pragma unroll
        for (int v = 0; v < 8; ++v) {
            int m = row0 + rm + v + 8 * g;
            h0[(size_t)m * H + nb + ln]      = fmaxf(acc0[v] + bv0, 0.f);
            h0[(size_t)m * H + nb + 16 + ln] = fmaxf(acc1[v] + bv1, 0.f);
        }
    } else {
#pragma unroll
        for (int v = 0; v < 8; ++v) {
            int m = row0 + rm + v + 8 * g;
            if (m < N) {
                h0[(size_t)m * H + nb + ln]      = fmaxf(acc0[v] + bv0, 0.f);
                h0[(size_t)m * H + nb + 16 + ln] = fmaxf(acc1[v] + bv1, 0.f);
            }
        }
    }
}

// ---------------------------------------------------------------------------
__global__ void k_zero(float* __restrict__ p, size_t n) {
    size_t i = (size_t)blockIdx.x * blockDim.x + threadIdx.x;
    if (i < n) p[i] = 0.f;
}

// ---------------------------------------------------------------------------
// Edge scatter: one wave per edge; float4 gather, f32 atomics (L2-resident)
// ---------------------------------------------------------------------------
__global__ __launch_bounds__(256) void k_scatter(const long long* __restrict__ ei,
                                                 const float* __restrict__ h, int hs,
                                                 float* __restrict__ agg,
                                                 float* __restrict__ deg, int E) {
    long long gid = (long long)blockIdx.x * blockDim.x + threadIdx.x;
    int e = (int)(gid >> 5);
    int lane = (int)(gid & 31);
    if (e >= E) return;
    int s = (int)ei[e];
    int d = (int)ei[(size_t)E + e];
    const float4 hv = *(const float4*)(h + (size_t)s * hs + lane * 4);
    float* ad = agg + (size_t)d * H + lane * 4;
    atomicAdd(ad + 0, hv.x);
    atomicAdd(ad + 1, hv.y);
    atomicAdd(ad + 2, hv.z);
    atomicAdd(ad + 3, hv.w);
    if (lane == 0) atomicAdd(deg + d, 1.0f);
}

// ---------------------------------------------------------------------------
// Fused layer: out = relu(LN(mean@Wl + bl + h@Wr)) + h
// K=256 WMMA GEMM; B halves (WlT then WrT) staged via TDM; LN epilogue.
// ---------------------------------------------------------------------------
__global__ __launch_bounds__(256) void k_layer(
    const float* __restrict__ agg, const float* __restrict__ deg,
    const float* __restrict__ hin, int hin_stride,
    const __bf16* __restrict__ BT,   // WlT | WrT, each BT_ELEMS
    const float* __restrict__ bl,
    const float* __restrict__ ln_g, const float* __restrict__ ln_b,
    float* __restrict__ out, int out_stride, int N) {
    __shared__ __bf16 sBt[BT_ELEMS];
    __shared__ __bf16 sA[2][32 * 40];
    __shared__ float  sC[32 * 128];
    __shared__ float  sRd[32];
    const int t = threadIdx.x;
    const int w = t >> 5, lane = t & 31;
    const int row0 = blockIdx.x * 32;
    const int rm = (w >> 2) * 16, nb = (w & 3) * 32;
    const int g = lane >> 4, ln = lane & 15;
    const bool full = (row0 + 32 <= N);

    if (t < 32) {
        int gm = row0 + t;
        sRd[t] = (gm < N) ? (1.0f / fmaxf(deg[gm], 1.0f)) : 0.f;
    }
#if HAVE_TDM
    if (w == 0) {
        tdm_load(BT, sBt, (unsigned)(BT_ELEMS * 2));
        __builtin_amdgcn_s_wait_tensorcnt(0);
    }
#else
    for (int i = t; i < BT_ELEMS / 8; i += 256)
        ((bf16x8*)sBt)[i] = ((const bf16x8*)BT)[i];
#endif

    f32x8 acc0 = {}, acc1 = {};
    for (int kb = 0; kb < 8; ++kb) {
        const bool left = kb < 4;
        const int ko = (kb & 3) * 32;
        if (kb == 4) {   // swap in WrT half
            __syncthreads();   // all frag reads of WlT complete (dscnt-waited)
#if HAVE_TDM
            if (w == 0) {
                tdm_load(BT + BT_ELEMS, sBt, (unsigned)(BT_ELEMS * 2));
                __builtin_amdgcn_s_wait_tensorcnt(0);
            }
#else
            for (int i = t; i < BT_ELEMS / 8; i += 256)
                ((bf16x8*)sBt)[i] = ((const bf16x8*)(BT + BT_ELEMS))[i];
#endif
        }
        {   // stage A chunk (mean for kb<4, h for kb>=4), double buffered
            int r = t >> 3, k4 = (t & 7) * 4;
            int gm = row0 + r;
            float4 v = make_float4(0.f, 0.f, 0.f, 0.f);
            if (gm < N) {
                if (left) {
                    v = *(const float4*)(agg + (size_t)gm * H + ko + k4);
                    float rd = sRd[r];
                    v.x *= rd; v.y *= rd; v.z *= rd; v.w *= rd;
                } else {
                    v = *(const float4*)(hin + (size_t)gm * hin_stride + ko + k4);
                }
            }
            *(bf16x4*)&sA[kb & 1][r * 40 + k4] = cvt4(v);
        }
        __syncthreads();
        bf16x16 a  = load_afrag(sA[kb & 1], rm + ln, g);
        bf16x16 b0 = load_bfrag(sBt, nb + ln, ko, g, 136);
        bf16x16 b1 = load_bfrag(sBt, nb + 16 + ln, ko, g, 136);
        acc0 = wmma_bf16(a, b0, acc0);
        acc1 = wmma_bf16(a, b1, acc1);
    }
    __syncthreads();
    // park (GEMM + bias) in LDS for the row-wise LayerNorm
    float bv0 = bl[nb + ln], bv1 = bl[nb + 16 + ln];
#pragma unroll
    for (int v = 0; v < 8; ++v) {
        int m = rm + v + 8 * g;
        sC[m * 128 + nb + ln]      = acc0[v] + bv0;
        sC[m * 128 + nb + 16 + ln] = acc1[v] + bv1;
    }
    __syncthreads();
    // LN + relu + residual: wave w owns rows 4w..4w+3; float4 per lane
    const float4 gg = *(const float4*)(ln_g + lane * 4);
    const float4 bb = *(const float4*)(ln_b + lane * 4);
    for (int r = 0; r < 4; ++r) {
        int row = w * 4 + r;
        int gm = row0 + row;
        float4 vv = *(const float4*)&sC[row * 128 + lane * 4];
        float s = vv.x + vv.y + vv.z + vv.w;
#pragma unroll
        for (int off = 16; off > 0; off >>= 1) s += __shfl_xor(s, off, 32);
        float mu = s * (1.f / 128.f);
        float dx = vv.x - mu, dy = vv.y - mu, dz = vv.z - mu, dw = vv.w - mu;
        float var = dx * dx + dy * dy + dz * dz + dw * dw;
#pragma unroll
        for (int off = 16; off > 0; off >>= 1) var += __shfl_xor(var, off, 32);
        float rs = rsqrtf(var * (1.f / 128.f) + 1e-5f);
        if (full || gm < N) {
            float4 hres = *(const float4*)(hin + (size_t)gm * hin_stride + lane * 4);
            float4 o;
            o.x = fmaxf(dx * rs * gg.x + bb.x, 0.f) + hres.x;
            o.y = fmaxf(dy * rs * gg.y + bb.y, 0.f) + hres.y;
            o.z = fmaxf(dz * rs * gg.z + bb.z, 0.f) + hres.z;
            o.w = fmaxf(dw * rs * gg.w + bb.w, 0.f) + hres.w;
            *(float4*)(out + (size_t)gm * out_stride + lane * 4) = o;
        }
    }
}

// ---------------------------------------------------------------------------
// Output projection: out = xs[N,384] @ W_out[384,16] + b_out
// ---------------------------------------------------------------------------
__global__ __launch_bounds__(256) void k_out_proj(const float* __restrict__ xs,
                                                  const __bf16* __restrict__ WoutT,
                                                  const float* __restrict__ bout,
                                                  float* __restrict__ out, int N) {
    __shared__ __bf16 sWt[16 * 392];   // transposed [n][k], col stride 392
    const int t = threadIdx.x;
    for (int i = t; i < (16 * 392) / 8; i += 256)
        ((bf16x8*)sWt)[i] = ((const bf16x8*)WoutT)[i];
    __syncthreads();
    const int w = t >> 5, lane = t & 31;
    const int g = lane >> 4, ln = lane & 15;
    const int row0 = blockIdx.x * 128 + w * 16;
    const int m = row0 + ln;
    f32x8 acc = {};
    for (int kb = 0; kb < 12; ++kb) {
        float4 x0 = make_float4(0.f, 0.f, 0.f, 0.f), x1 = x0, x2 = x0, x3 = x0;
        if (m < N) {
            const float* p = xs + (size_t)m * (3 * H) + kb * 32 + g * 8;
            x0 = *(const float4*)(p);
            x1 = *(const float4*)(p + 4);
            x2 = *(const float4*)(p + 16);
            x3 = *(const float4*)(p + 20);
        }
        bf16x8 alo, ahi;
        alo[0] = (__bf16)x0.x; alo[1] = (__bf16)x0.y; alo[2] = (__bf16)x0.z;
        alo[3] = (__bf16)x0.w; alo[4] = (__bf16)x1.x; alo[5] = (__bf16)x1.y;
        alo[6] = (__bf16)x1.z; alo[7] = (__bf16)x1.w;
        ahi[0] = (__bf16)x2.x; ahi[1] = (__bf16)x2.y; ahi[2] = (__bf16)x2.z;
        ahi[3] = (__bf16)x2.w; ahi[4] = (__bf16)x3.x; ahi[5] = (__bf16)x3.y;
        ahi[6] = (__bf16)x3.z; ahi[7] = (__bf16)x3.w;
        bf16x16 a = CAT16(alo, ahi);
        bf16x16 b = load_bfrag(sWt, ln, kb * 32, g, 392);
        acc = wmma_bf16(a, b, acc);
    }
    float bv = bout[ln];
#pragma unroll
    for (int v = 0; v < 8; ++v) {
        int mm = row0 + v + 8 * g;
        if (mm < N) out[(size_t)mm * COUT + ln] = acc[v] + bv;
    }
}

// ---------------------------------------------------------------------------
extern "C" void kernel_launch(void* const* d_in, const int* in_sizes, int n_in,
                              void* d_out, int out_size, void* d_ws, size_t ws_size,
                              hipStream_t stream) {
    (void)n_in; (void)out_size; (void)ws_size;
    const float*     x    = (const float*)d_in[0];
    const long long* ei   = (const long long*)d_in[1];
    const float*     Win  = (const float*)d_in[2];
    const float*     bin  = (const float*)d_in[3];
    const float*     Wl   = (const float*)d_in[4];
    const float*     bl   = (const float*)d_in[5];
    const float*     Wr   = (const float*)d_in[6];
    const float*     lng  = (const float*)d_in[7];
    const float*     lnb  = (const float*)d_in[8];
    const float*     Wout = (const float*)d_in[9];
    const float*     bout = (const float*)d_in[10];
    const int N = in_sizes[0] / H;
    const int E = in_sizes[1] / 2;

    // workspace layout: h0 | xs | agg | deg | WinT | WlrT[3] | WoutT
    char* ws = (char*)d_ws;
    size_t o = 0;
    float* h0  = (float*)(ws + o); o += (size_t)N * H * 4;
    float* xs  = (float*)(ws + o); o += (size_t)N * 3 * H * 4;
    float* agg = (float*)(ws + o); o += (size_t)N * H * 4;
    float* deg = (float*)(ws + o); o += (size_t)N * 4;
    o = (o + 15) & ~(size_t)15;
    __bf16* WinT  = (__bf16*)(ws + o); o += (size_t)BT_ELEMS * 2;
    __bf16* WlrT  = (__bf16*)(ws + o); o += (size_t)LAYERS * 2 * BT_ELEMS * 2;
    __bf16* WoutT = (__bf16*)(ws + o);
    float* out = (float*)d_out;

    // ---- weight prep (tiny): bf16 transposed+padded images ----
    k_transpose<<<(128 * 128 + 255) / 256, 256, 0, stream>>>(Win, WinT, 128, 128, 136);
    for (int l = 0; l < LAYERS; ++l) {
        k_transpose<<<(128 * 128 + 255) / 256, 256, 0, stream>>>(
            Wl + (size_t)l * H * H, WlrT + (size_t)(2 * l) * BT_ELEMS, 128, 128, 136);
        k_transpose<<<(128 * 128 + 255) / 256, 256, 0, stream>>>(
            Wr + (size_t)l * H * H, WlrT + (size_t)(2 * l + 1) * BT_ELEMS, 128, 128,
            136);
    }
    k_transpose<<<(384 * 16 + 255) / 256, 256, 0, stream>>>(Wout, WoutT, 384, 16, 392);

    const int mtiles = (N + 31) / 32;
    k_in_proj<<<mtiles, 256, 0, stream>>>(x, WinT, bin, h0, N);

    for (int l = 0; l < LAYERS; ++l) {
        size_t zn = (size_t)N * H + (size_t)N;   // agg + deg are contiguous
        k_zero<<<(int)((zn + 255) / 256), 256, 0, stream>>>(agg, zn);
        const float* hin = (l == 0) ? h0 : (xs + (size_t)(l - 1) * H);
        int hs = (l == 0) ? H : 3 * H;
        long long sthreads = (long long)E * 32;
        k_scatter<<<(int)((sthreads + 255) / 256), 256, 0, stream>>>(ei, hin, hs, agg,
                                                                     deg, E);
        k_layer<<<mtiles, 256, 0, stream>>>(
            agg, deg, hin, hs, WlrT + (size_t)(2 * l) * BT_ELEMS, bl + (size_t)l * H,
            lng + (size_t)l * H, lnb + (size_t)l * H, xs + (size_t)l * H, 3 * H, N);
    }
    k_out_proj<<<(N + 127) / 128, 256, 0, stream>>>(xs, Wout ? WoutT : WoutT, bout, out,
                                                    N);
}
